// EdgeAwareGATFusion_32581621907893
// MI455X (gfx1250) — matrix operand order repaired and checked by reading, not compiled, not run on hardware
//
#include <hip/hip_runtime.h>

// ---------- types ----------
typedef __bf16 bf16_t;
typedef bf16_t v16bf __attribute__((ext_vector_type(16)));
typedef bf16_t v8bf  __attribute__((ext_vector_type(8)));
typedef float  v8f   __attribute__((ext_vector_type(8)));

#define DM   256
#define DFF  512
#define KMEM 768

// ---------- CDNA5 async global->LDS copy (16B per lane) ----------
__device__ __forceinline__ void async_copy16(void* lds, const void* gptr) {
    unsigned l = (unsigned)(uintptr_t)lds;        // LDS address = low 32 bits
    asm volatile("global_load_async_to_lds_b128 %0, %1, off"
                 :: "v"(l), "v"(gptr) : "memory");
}
__device__ __forceinline__ void wait_async() {
    asm volatile("s_wait_asynccnt 0x0" ::: "memory");
}

// ---------- WMMA fragment helpers ----------
__device__ __forceinline__ v16bf lda_row(const bf16_t* rowp, int k) {
    v8bf lo = *(const v8bf*)(rowp + k);
    v8bf hi = *(const v8bf*)(rowp + k + 16);
    v16bf r;
#pragma unroll
    for (int i = 0; i < 8; ++i) { r[i] = lo[i]; r[i + 8] = hi[i]; }
    return r;
}

__device__ __forceinline__ v16bf ldb_wt(const bf16_t* WT, int K, int n0, int k0, int lane) {
    const bf16_t* p = WT + (size_t)(n0 + (lane & 15)) * K + k0 + ((lane & 16) ? 8 : 0);
    v8bf lo = *(const v8bf*)p;
    v8bf hi = *(const v8bf*)(p + 16);
    v16bf r;
#pragma unroll
    for (int i = 0; i < 8; ++i) { r[i] = lo[i]; r[i + 8] = hi[i]; }
    return r;
}

// MW x NW register-blocked wave GEMM. A panel in LDS (LDA elements/row),
// m-tiles at rows {0,16,..}, n-tiles at cols {n0, n0+nstride, ...}.
template <int K, int LDA, int MW, int NW>
__device__ __forceinline__ void gemm_block(const bf16_t* sA, const bf16_t* WT,
                                           int n0, int nstride, int lane,
                                           v8f acc[MW][NW]) {
    const int ka = (lane & 16) ? 8 : 0;
    const int mrow = lane & 15;
#pragma unroll
    for (int k0 = 0; k0 < K; k0 += 32) {
        v16bf a[MW];
#pragma unroll
        for (int i = 0; i < MW; ++i)
            a[i] = lda_row(sA + (size_t)(i * 16 + mrow) * LDA, k0 + ka);
#pragma unroll
        for (int j = 0; j < NW; ++j) {
            v16bf b = ldb_wt(WT, K, n0 + j * nstride, k0, lane);
#pragma unroll
            for (int i = 0; i < MW; ++i)
                acc[i][j] = __builtin_amdgcn_wmma_f32_16x16x32_bf16(
                    false, a[i], false, b, (short)0, acc[i][j], false, false);
        }
    }
}

template <int K>
__device__ __forceinline__ v8f wave_gemm(const bf16_t* aRow, const bf16_t* WT, int n0, int lane) {
    const int ka = (lane & 16) ? 8 : 0;
    v8f acc = {};
#pragma unroll
    for (int k0 = 0; k0 < K; k0 += 32) {
        v16bf a = lda_row(aRow, k0 + ka);
        v16bf b = ldb_wt(WT, K, n0, k0, lane);
        acc = __builtin_amdgcn_wmma_f32_16x16x32_bf16(false, a, false, b, (short)0, acc, false, false);
    }
    return acc;
}

__device__ __forceinline__ float wave_red_add(float v) {
#pragma unroll
    for (int off = 16; off > 0; off >>= 1) v += __shfl_xor(v, off, 32);
    return v;
}

__device__ __forceinline__ float half_red_add(float v) {  // reduce within 16-lane group
#pragma unroll
    for (int off = 8; off > 0; off >>= 1) v += __shfl_xor(v, off, 32);
    return v;
}

__device__ __forceinline__ void atomicMaxF(float* addr, float val) {
    unsigned int* ua = (unsigned int*)addr;
    unsigned int old = *ua;
    for (;;) {
        float f = __uint_as_float(old);
        if (f >= val) break;
        unsigned int assumed = old;
        old = atomicCAS(ua, assumed, __float_as_uint(val));
        if (old == assumed) break;
    }
}

// ---------- small utility kernels ----------
__global__ void k_tcvt(const float* __restrict__ src, bf16_t* __restrict__ dst, int K, int Nout) {
    int idx = blockIdx.x * 256 + threadIdx.x;
    if (idx >= K * Nout) return;
    int k = idx / Nout, n = idx - k * Nout;
    dst[(size_t)n * K + k] = (bf16_t)src[idx];
}

__global__ void k_cvt(const float* __restrict__ s, bf16_t* __restrict__ d, int n) {
    int i = blockIdx.x * 256 + threadIdx.x;
    if (i < n) d[i] = (bf16_t)s[i];
}

__global__ void k_init(float* aggr, float* smax, float* sden, int n256, int n8) {
    int i = blockIdx.x * 256 + threadIdx.x;
    if (i < n256) aggr[i] = 0.f;
    if (i < n8) { smax[i] = -3.0e38f; sden[i] = 0.f; }
}

// ---------- edge memory MLP: mem = relu(ln(cat(x_i,x_j,ea) @ W_mem + b)) ----------
__global__ __launch_bounds__(256) void k_mem_gemm(
    const bf16_t* __restrict__ xb, const float* __restrict__ edge_attr,
    const int* __restrict__ srcI, const int* __restrict__ dstI,
    const bf16_t* __restrict__ WT, const float* __restrict__ bias,
    const float* __restrict__ gam, const float* __restrict__ bet,
    bf16_t* __restrict__ memb) {
    constexpr int LDA = KMEM + 8;
    __shared__ __align__(16) char smem[32 * LDA * 2];      // 49.7KB; aliased with sC (32KB)
    bf16_t* sA = (bf16_t*)smem;
    float*  sC = (float*)smem;
    __shared__ int sSrc[32], sDst[32];
    const int tid = threadIdx.x;
    const size_t e0 = (size_t)blockIdx.x * 32;
    if (tid < 32) { sSrc[tid] = srcI[e0 + tid]; sDst[tid] = dstI[e0 + tid]; }
    __syncthreads();
    // x_i | x_j : pure bf16 copies -> async DMA to LDS (32 rows x 64 chunks of 16B)
    for (int c = tid; c < 32 * 64; c += 256) {
        int row = c >> 6, ch = c & 63;
        const bf16_t* srcp = (ch < 32) ? (xb + (size_t)sDst[row] * DM + ch * 8)
                                       : (xb + (size_t)sSrc[row] * DM + (ch - 32) * 8);
        async_copy16(sA + (size_t)row * LDA + ch * 8, srcp);
    }
    // edge_attr : f32 -> bf16 conversion on the VALU path
    for (int idx = tid; idx < 32 * DM; idx += 256) {
        int row = idx >> 8, col = idx & (DM - 1);
        sA[(size_t)row * LDA + 2 * DM + col] = (bf16_t)edge_attr[(e0 + row) * DM + col];
    }
    wait_async();
    __syncthreads();
    const int wave = tid >> 5, lane = tid & 31;
    const int rbase = (lane & 16) ? 8 : 0;
    const int n0 = wave * 16;
    v8f acc[2][2] = {};
    gemm_block<KMEM, LDA, 2, 2>(sA, WT, n0, 128, lane, acc);
    __syncthreads();                       // everyone done reading sA
#pragma unroll
    for (int j = 0; j < 2; ++j) {
        const int col = n0 + j * 128 + (lane & 15);
        const float bv = bias[col];
#pragma unroll
        for (int i = 0; i < 2; ++i)
#pragma unroll
            for (int r = 0; r < 8; ++r)
                sC[(size_t)(i * 16 + rbase + r) * DM + col] = acc[i][j][r] + bv;
    }
    __syncthreads();
#pragma unroll
    for (int rr = 0; rr < 4; ++rr) {
        const int row = wave * 4 + rr;
        float vals[8], s = 0.f, s2 = 0.f;
#pragma unroll
        for (int j = 0; j < 8; ++j) { float v = sC[(size_t)row * DM + lane + 32 * j]; vals[j] = v; s += v; s2 += v * v; }
        s = wave_red_add(s); s2 = wave_red_add(s2);
        const float mean = s * (1.f / DM);
        const float inv  = rsqrtf(s2 * (1.f / DM) - mean * mean + 1e-5f);
        const size_t base = (e0 + row) * DM;
#pragma unroll
        for (int j = 0; j < 8; ++j) {
            const int col = lane + 32 * j;
            float v = (vals[j] - mean) * inv * gam[col] + bet[col];
            memb[base + col] = (bf16_t)(v > 0.f ? v : 0.f);
        }
    }
}

// ---------- edge update: edge_out = ln(ea + relu(ln(mem@W_eu+b))) ----------
__global__ __launch_bounds__(256) void k_edge_update(
    const bf16_t* __restrict__ memb, const float* __restrict__ edge_attr,
    const bf16_t* __restrict__ WT, const float* __restrict__ bias,
    const float* __restrict__ g_eu, const float* __restrict__ be_eu,
    const float* __restrict__ g_en, const float* __restrict__ be_en,
    float* __restrict__ out_edge) {
    constexpr int LDA = DM + 8;
    __shared__ __align__(16) char smem[32 * DM * 4];       // 32KB >= max(A panel 16.9KB, C 32KB)
    bf16_t* sA = (bf16_t*)smem;
    float*  sC = (float*)smem;
    const int tid = threadIdx.x;
    const size_t e0 = (size_t)blockIdx.x * 32;
    for (int c = tid; c < 32 * 32; c += 256) {             // 32 rows x 32 chunks of 16B
        int row = c >> 5, ch = c & 31;
        async_copy16(sA + (size_t)row * LDA + ch * 8, memb + (e0 + row) * DM + ch * 8);
    }
    wait_async();
    __syncthreads();
    const int wave = tid >> 5, lane = tid & 31;
    const int rbase = (lane & 16) ? 8 : 0;
    const int n0 = wave * 16;
    v8f acc[2][2] = {};
    gemm_block<DM, LDA, 2, 2>(sA, WT, n0, 128, lane, acc);
    __syncthreads();
#pragma unroll
    for (int j = 0; j < 2; ++j) {
        const int col = n0 + j * 128 + (lane & 15);
        const float bv = bias[col];
#pragma unroll
        for (int i = 0; i < 2; ++i)
#pragma unroll
            for (int r = 0; r < 8; ++r)
                sC[(size_t)(i * 16 + rbase + r) * DM + col] = acc[i][j][r] + bv;
    }
    __syncthreads();
#pragma unroll
    for (int rr = 0; rr < 4; ++rr) {
        const int row = wave * 4 + rr;
        const size_t base = (e0 + row) * DM;
        float vals[8], s = 0.f, s2 = 0.f;
#pragma unroll
        for (int j = 0; j < 8; ++j) { float v = sC[(size_t)row * DM + lane + 32 * j]; vals[j] = v; s += v; s2 += v * v; }
        s = wave_red_add(s); s2 = wave_red_add(s2);
        float mean = s * (1.f / DM);
        float inv  = rsqrtf(s2 * (1.f / DM) - mean * mean + 1e-5f);
        s = 0.f; s2 = 0.f;
#pragma unroll
        for (int j = 0; j < 8; ++j) {
            const int col = lane + 32 * j;
            float d = (vals[j] - mean) * inv * g_eu[col] + be_eu[col];
            d = d > 0.f ? d : 0.f;
            float v = edge_attr[base + col] + d;
            vals[j] = v; s += v; s2 += v * v;
        }
        s = wave_red_add(s); s2 = wave_red_add(s2);
        mean = s * (1.f / DM);
        inv  = rsqrtf(s2 * (1.f / DM) - mean * mean + 1e-5f);
#pragma unroll
        for (int j = 0; j < 8; ++j) {
            const int col = lane + 32 * j;
            out_edge[base + col] = (vals[j] - mean) * inv * g_en[col] + be_en[col];
        }
    }
}

// ---------- q/k/v projections + per-head logits (head-aligned waves) ----------
__global__ __launch_bounds__(256) void k_qkv(
    const bf16_t* __restrict__ xb, const bf16_t* __restrict__ memb,
    const int* __restrict__ dstI,
    const bf16_t* __restrict__ WTq, const bf16_t* __restrict__ WTk, const bf16_t* __restrict__ WTv,
    bf16_t* __restrict__ vb, float* __restrict__ logits) {
    constexpr int LDA = DM + 8;
    __shared__ bf16_t sXi[32 * LDA];
    __shared__ bf16_t sM [32 * LDA];
    __shared__ int    sDst[32];
    const int tid = threadIdx.x;
    const size_t e0 = (size_t)blockIdx.x * 32;
    if (tid < 32) sDst[tid] = dstI[e0 + tid];
    __syncthreads();
    for (int c = tid; c < 32 * 32; c += 256) {             // 32 rows x 32 chunks of 16B
        int row = c >> 5, ch = c & 31;
        async_copy16(sXi + (size_t)row * LDA + ch * 8, xb + (size_t)sDst[row] * DM + ch * 8);
        async_copy16(sM  + (size_t)row * LDA + ch * 8, memb + (e0 + row) * DM + ch * 8);
    }
    wait_async();
    __syncthreads();
    const int wave = tid >> 5, lane = tid & 31;     // wave == head
    const int rbase = (lane & 16) ? 8 : 0;
    const int n0 = wave * 32;                       // head h covers cols [32h, 32h+32)
    v8f qa[2][2] = {}, ka_[2][2] = {};
    gemm_block<DM, LDA, 2, 2>(sXi, WTq, n0, 16, lane, qa);
    gemm_block<DM, LDA, 2, 2>(sM,  WTk, n0, 16, lane, ka_);
#pragma unroll
    for (int i = 0; i < 2; ++i) {
#pragma unroll
        for (int r = 0; r < 8; ++r) {
            float p = qa[i][0][r] * ka_[i][0][r] + qa[i][1][r] * ka_[i][1][r];
            p = half_red_add(p);                    // sum over 16 lanes of this half
            if ((lane & 15) == 0) {
                const int mrow = i * 16 + rbase + r;
                logits[(e0 + mrow) * 8 + wave] = p * 0.17677669529663689f;  // 1/sqrt(32)
            }
        }
    }
    v8f va[2][2] = {};
    gemm_block<DM, LDA, 2, 2>(sM, WTv, n0, 16, lane, va);
#pragma unroll
    for (int j = 0; j < 2; ++j) {
        const int col = n0 + j * 16 + (lane & 15);
#pragma unroll
        for (int i = 0; i < 2; ++i)
#pragma unroll
            for (int r = 0; r < 8; ++r)
                vb[(e0 + i * 16 + rbase + r) * DM + col] = (bf16_t)va[i][j][r];
    }
}

// ---------- segment softmax pieces ----------
__global__ void k_segmax(const float* __restrict__ logits, const int* __restrict__ dstI,
                         float* __restrict__ smax, int EH) {
    int i = blockIdx.x * 256 + threadIdx.x;
    if (i >= EH) return;
    int e = i >> 3, h = i & 7;
    atomicMaxF(&smax[(size_t)dstI[e] * 8 + h], logits[i]);
}

__global__ void k_expsum(float* __restrict__ exl, const int* __restrict__ dstI,
                         const float* __restrict__ smax, float* __restrict__ sden, int EH) {
    int i = blockIdx.x * 256 + threadIdx.x;
    if (i >= EH) return;
    int e = i >> 3, h = i & 7;
    int d = dstI[e];
    float v = expf(exl[i] - smax[(size_t)d * 8 + h]);
    exl[i] = v;
    atomicAdd(&sden[(size_t)d * 8 + h], v);
}

// ---------- msg = (attn*v)@Wo, scatter-add to aggr ----------
__global__ __launch_bounds__(256) void k_msg(
    const bf16_t* __restrict__ vb, const float* __restrict__ exl,
    const float* __restrict__ sden, const int* __restrict__ dstI,
    const bf16_t* __restrict__ WTo, float* __restrict__ aggr) {
    constexpr int LDA = DM + 8;
    __shared__ bf16_t sA[32 * LDA];
    __shared__ int sDst[32];
    const int tid = threadIdx.x;
    const size_t e0 = (size_t)blockIdx.x * 32;
    if (tid < 32) sDst[tid] = dstI[e0 + tid];
    __syncthreads();
    for (int idx = tid; idx < 32 * DM; idx += 256) {
        int row = idx >> 8, col = idx & (DM - 1), h = col >> 5;
        size_t e = e0 + row;
        float a = exl[e * 8 + h] / (sden[(size_t)sDst[row] * 8 + h] + 1e-16f);
        sA[(size_t)row * LDA + col] = (bf16_t)((float)vb[e * DM + col] * a);
    }
    __syncthreads();
    const int wave = tid >> 5, lane = tid & 31;
    const int rbase = (lane & 16) ? 8 : 0;
    const int n0 = wave * 16;
    v8f acc[2][2] = {};
    gemm_block<DM, LDA, 2, 2>(sA, WTo, n0, 128, lane, acc);
#pragma unroll
    for (int j = 0; j < 2; ++j) {
        const int col = n0 + j * 128 + (lane & 15);
#pragma unroll
        for (int i = 0; i < 2; ++i)
#pragma unroll
            for (int r = 0; r < 8; ++r)
                atomicAdd(&aggr[(size_t)sDst[i * 16 + rbase + r] * DM + col], acc[i][j][r]);
    }
}

// ---------- h1 = ln(x + aggr) ----------
__global__ __launch_bounds__(256) void k_h1(
    const float* __restrict__ x, const float* __restrict__ aggr,
    const float* __restrict__ g1, const float* __restrict__ be1,
    float* __restrict__ h1f, bf16_t* __restrict__ h1b) {
    const int tid = threadIdx.x, wave = tid >> 5, lane = tid & 31;
    const size_t n0 = (size_t)blockIdx.x * 16;
#pragma unroll
    for (int rr = 0; rr < 2; ++rr) {
        const size_t row = n0 + wave * 2 + rr;
        const size_t base = row * DM;
        float vals[8], s = 0.f, s2 = 0.f;
#pragma unroll
        for (int j = 0; j < 8; ++j) {
            const int col = lane + 32 * j;
            float v = x[base + col] + aggr[base + col];
            vals[j] = v; s += v; s2 += v * v;
        }
        s = wave_red_add(s); s2 = wave_red_add(s2);
        const float mean = s * (1.f / DM);
        const float inv  = rsqrtf(s2 * (1.f / DM) - mean * mean + 1e-5f);
#pragma unroll
        for (int j = 0; j < 8; ++j) {
            const int col = lane + 32 * j;
            float v = (vals[j] - mean) * inv * g1[col] + be1[col];
            h1f[base + col] = v;
            h1b[base + col] = (bf16_t)v;
        }
    }
}

// ---------- ffn1 = relu(h1 @ W_f1 + b_f1), N x 512 ----------
__global__ __launch_bounds__(256) void k_ffn1(
    const bf16_t* __restrict__ h1b, const bf16_t* __restrict__ WT,
    const float* __restrict__ bias, bf16_t* __restrict__ f1b) {
    constexpr int LDA = DM + 8;
    __shared__ bf16_t sA[16 * LDA];
    const int tid = threadIdx.x;
    const size_t n0r = (size_t)blockIdx.x * 16;
    for (int c = tid; c < 16 * 32; c += 256) {             // 16 rows x 32 chunks of 16B
        int row = c >> 5, ch = c & 31;
        async_copy16(sA + (size_t)row * LDA + ch * 8, h1b + (n0r + row) * DM + ch * 8);
    }
    wait_async();
    __syncthreads();
    const int wave = tid >> 5, lane = tid & 31;
    const int rbase = (lane & 16) ? 8 : 0;
    const bf16_t* aRow = sA + (size_t)(lane & 15) * LDA;
#pragma unroll
    for (int t = 0; t < 4; ++t) {
        const int n0 = (wave + 8 * t) * 16;
        v8f acc = wave_gemm<DM>(aRow, WT, n0, lane);
        const int col = n0 + (lane & 15);
        const float bv = bias[col];
#pragma unroll
        for (int r = 0; r < 8; ++r) {
            float v = acc[r] + bv;
            f1b[(n0r + rbase + r) * DFF + col] = (bf16_t)(v > 0.f ? v : 0.f);
        }
    }
}

// ---------- out = ln(h1 + ffn1 @ W_f2 + b_f2) ----------
__global__ __launch_bounds__(256) void k_ffn2(
    const bf16_t* __restrict__ f1b, const bf16_t* __restrict__ WT,
    const float* __restrict__ bias, const float* __restrict__ h1f,
    const float* __restrict__ g2, const float* __restrict__ be2,
    float* __restrict__ out) {
    constexpr int LDA = DFF + 8;
    __shared__ bf16_t sA[16 * LDA];
    __shared__ float  sC[16][DM];
    const int tid = threadIdx.x;
    const size_t n0r = (size_t)blockIdx.x * 16;
    for (int c = tid; c < 16 * 64; c += 256) {             // 16 rows x 64 chunks of 16B
        int row = c >> 6, ch = c & 63;
        async_copy16(sA + (size_t)row * LDA + ch * 8, f1b + (n0r + row) * DFF + ch * 8);
    }
    wait_async();
    __syncthreads();
    const int wave = tid >> 5, lane = tid & 31;
    const int rbase = (lane & 16) ? 8 : 0;
    const bf16_t* aRow = sA + (size_t)(lane & 15) * LDA;
#pragma unroll
    for (int t = 0; t < 2; ++t) {
        const int n0 = (wave + 8 * t) * 16;
        v8f acc = wave_gemm<DFF>(aRow, WT, n0, lane);
        const int col = n0 + (lane & 15);
        const float bv = bias[col];
#pragma unroll
        for (int r = 0; r < 8; ++r) sC[rbase + r][col] = acc[r] + bv;
    }
    __syncthreads();
#pragma unroll
    for (int rr = 0; rr < 2; ++rr) {
        const int row = wave * 2 + rr;
        const size_t base = (n0r + row) * DM;
        float vals[8], s = 0.f, s2 = 0.f;
#pragma unroll
        for (int j = 0; j < 8; ++j) {
            const int col = lane + 32 * j;
            float v = h1f[base + col] + sC[row][col];
            vals[j] = v; s += v; s2 += v * v;
        }
        s = wave_red_add(s); s2 = wave_red_add(s2);
        const float mean = s * (1.f / DM);
        const float inv  = rsqrtf(s2 * (1.f / DM) - mean * mean + 1e-5f);
#pragma unroll
        for (int j = 0; j < 8; ++j) {
            const int col = lane + 32 * j;
            out[base + col] = (vals[j] - mean) * inv * g2[col] + be2[col];
        }
    }
}

// ---------- host launch ----------
extern "C" void kernel_launch(void* const* d_in, const int* in_sizes, int n_in,
                              void* d_out, int out_size, void* d_ws, size_t ws_size,
                              hipStream_t stream) {
    (void)n_in; (void)out_size; (void)ws_size;
    const float* x     = (const float*)d_in[0];
    const int*   ei    = (const int*)d_in[1];
    const float* ea    = (const float*)d_in[2];
    const float* W_mem = (const float*)d_in[3];
    const float* b_mem = (const float*)d_in[4];
    const float* g_mem = (const float*)d_in[5];
    const float* be_mem= (const float*)d_in[6];
    const float* Wq    = (const float*)d_in[7];
    const float* Wk    = (const float*)d_in[8];
    const float* Wv    = (const float*)d_in[9];
    const float* Wo    = (const float*)d_in[10];
    const float* W_eu  = (const float*)d_in[11];
    const float* b_eu  = (const float*)d_in[12];
    const float* g_eu  = (const float*)d_in[13];
    const float* be_eu = (const float*)d_in[14];
    const float* g_en  = (const float*)d_in[15];
    const float* be_en = (const float*)d_in[16];
    const float* W_f1  = (const float*)d_in[17];
    const float* b_f1  = (const float*)d_in[18];
    const float* W_f2  = (const float*)d_in[19];
    const float* b_f2  = (const float*)d_in[20];
    const float* g1    = (const float*)d_in[21];
    const float* be1   = (const float*)d_in[22];
    const float* g2    = (const float*)d_in[23];
    const float* be2   = (const float*)d_in[24];

    const int N = in_sizes[0] / DM;
    const int E = in_sizes[1] / 2;
    const int* srcI = ei;
    const int* dstI = ei + E;

    // workspace carve-up
    char* w = (char*)d_ws;
    bf16_t* WT_mem = (bf16_t*)w; w += (size_t)DM * KMEM * 2;
    bf16_t* WT_eu  = (bf16_t*)w; w += (size_t)DM * DM * 2;
    bf16_t* WT_q   = (bf16_t*)w; w += (size_t)DM * DM * 2;
    bf16_t* WT_k   = (bf16_t*)w; w += (size_t)DM * DM * 2;
    bf16_t* WT_v   = (bf16_t*)w; w += (size_t)DM * DM * 2;
    bf16_t* WT_o   = (bf16_t*)w; w += (size_t)DM * DM * 2;
    bf16_t* WT_f1  = (bf16_t*)w; w += (size_t)DFF * DM * 2;
    bf16_t* WT_f2  = (bf16_t*)w; w += (size_t)DM * DFF * 2;
    bf16_t* xb     = (bf16_t*)w; w += (size_t)N * DM * 2;
    bf16_t* memb   = (bf16_t*)w; w += (size_t)E * DM * 2;
    bf16_t* vb     = (bf16_t*)w; w += (size_t)E * DM * 2;
    float*  exl    = (float*)w;  w += (size_t)E * 8 * 4;
    float*  smax   = (float*)w;  w += (size_t)N * 8 * 4;
    float*  sden   = (float*)w;  w += (size_t)N * 8 * 4;
    float*  aggr   = (float*)w;  w += (size_t)N * DM * 4;
    float*  h1f    = (float*)w;  w += (size_t)N * DM * 4;
    bf16_t* h1b    = (bf16_t*)w; w += (size_t)N * DM * 2;
    bf16_t* f1b    = (bf16_t*)w; w += (size_t)N * DFF * 2;

    float* out_h    = (float*)d_out;
    float* out_edge = out_h + (size_t)N * DM;

    auto grid1 = [](int n) { return (n + 255) / 256; };
    k_tcvt<<<grid1(KMEM * DM), 256, 0, stream>>>(W_mem, WT_mem, KMEM, DM);
    k_tcvt<<<grid1(DM * DM),   256, 0, stream>>>(W_eu,  WT_eu,  DM, DM);
    k_tcvt<<<grid1(DM * DM),   256, 0, stream>>>(Wq,    WT_q,   DM, DM);
    k_tcvt<<<grid1(DM * DM),   256, 0, stream>>>(Wk,    WT_k,   DM, DM);
    k_tcvt<<<grid1(DM * DM),   256, 0, stream>>>(Wv,    WT_v,   DM, DM);
    k_tcvt<<<grid1(DM * DM),   256, 0, stream>>>(Wo,    WT_o,   DM, DM);
    k_tcvt<<<grid1(DM * DFF),  256, 0, stream>>>(W_f1,  WT_f1,  DM, DFF);
    k_tcvt<<<grid1(DFF * DM),  256, 0, stream>>>(W_f2,  WT_f2,  DFF, DM);
    k_cvt<<<grid1(N * DM), 256, 0, stream>>>(x, xb, N * DM);
    k_init<<<grid1(N * DM), 256, 0, stream>>>(aggr, smax, sden, N * DM, N * 8);

    const int gridE = E / 32;   // M_TILE = 32 on edge kernels
    const int gridN = N / 16;

    k_mem_gemm<<<gridE, 256, 0, stream>>>(xb, ea, srcI, dstI, WT_mem, b_mem, g_mem, be_mem, memb);
    k_edge_update<<<gridE, 256, 0, stream>>>(memb, ea, WT_eu, b_eu, g_eu, be_eu, g_en, be_en, out_edge);
    k_qkv<<<gridE, 256, 0, stream>>>(xb, memb, dstI, WT_q, WT_k, WT_v, vb, exl);
    k_segmax<<<grid1(E * 8), 256, 0, stream>>>(exl, dstI, smax, E * 8);
    k_expsum<<<grid1(E * 8), 256, 0, stream>>>(exl, dstI, smax, sden, E * 8);
    k_msg<<<gridE, 256, 0, stream>>>(vb, exl, sden, dstI, WT_o, aggr);
    k_h1<<<gridN, 256, 0, stream>>>(x, aggr, g1, be1, h1f, h1b);
    k_ffn1<<<gridN, 256, 0, stream>>>(h1b, WT_f1, b_f1, f1b);
    k_ffn2<<<gridN, 256, 0, stream>>>(f1b, WT_f2, b_f2, h1f, g2, be2, out_h);
}
